// AttentionOp_146028888400
// MI455X (gfx1250) — matrix operand order repaired and checked by reading, not compiled, not run on hardware
//
#include <hip/hip_runtime.h>
#include <hip/hip_bf16.h>
#include <float.h>
#include <limits.h>

typedef __attribute__((ext_vector_type(16))) _Float16 v16h;
typedef __attribute__((ext_vector_type(8)))  float    v8f;

#define BB    2
#define SEQ   2048
#define HQN   16
#define HKVN  8
#define DD    128
#define GRP   (HQN / HKVN)      // 2
#define QTS   16                // queries per wave tile
#define KTS   32                // keys per inner iteration
#define NQT   (SEQ / QTS)       // 128
#define NKT   (SEQ / KTS)       // 64
#define PPAD  40                // padded row stride (halves) for P transpose buffer
#define MASKV (-0.7f * 3.402823466e38f)

// ---------------------------------------------------------------------------
// Prologue: compute block labels idx[b][s] = m * cumsum(boundary), and per
// 32-key-tile nonzero-label [min,max] ranges for exact tile skipping.
// One block per batch, 256 threads, 8 elements per thread, Hillis-Steele scan.
// ---------------------------------------------------------------------------
__global__ void mask_scan_kernel(const int* __restrict__ bm,
                                 int* __restrict__ idx,
                                 int* __restrict__ ktmin,
                                 int* __restrict__ ktmax) {
  __shared__ int part[256];
  __shared__ int sidx[SEQ];
  const int b = blockIdx.x;
  const int t = threadIdx.x;
  const int* m = bm + b * SEQ;
  const int base = t * 8;

  int mv[8], bnd[8];
  const int prev = (base == 0) ? 0 : m[base - 1];
  int cnt = 0;
#pragma unroll
  for (int i = 0; i < 8; ++i) {
    mv[i]  = m[base + i];
    bnd[i] = (mv[i] > ((i == 0) ? prev : mv[i - 1])) ? 1 : 0;
    cnt += bnd[i];
  }
  part[t] = cnt;
  __syncthreads();
  for (int off = 1; off < 256; off <<= 1) {
    const int vv = (t >= off) ? part[t - off] : 0;
    __syncthreads();
    part[t] += vv;
    __syncthreads();
  }
  int c = part[t] - cnt;  // exclusive prefix of boundary count
#pragma unroll
  for (int i = 0; i < 8; ++i) {
    c += bnd[i];
    const int vi = mv[i] ? c : 0;
    sidx[base + i] = vi;
    idx[b * SEQ + base + i] = vi;
  }
  __syncthreads();
  if (t < NKT) {
    int lo = INT_MAX, hh = 0;
    for (int i = 0; i < KTS; ++i) {
      const int vv = sidx[t * KTS + i];
      if (vv > 0) { lo = min(lo, vv); hh = max(hh, vv); }
    }
    ktmin[b * NKT + t] = lo;
    ktmax[b * NKT + t] = hh;
  }
}

// ---------------------------------------------------------------------------
// Flash attention: one wave (32 threads) per block; each wave produces a
// 16-query x 128-D output tile for one (b, h). Online softmax, WMMA f16.
// ---------------------------------------------------------------------------
__launch_bounds__(32)
__global__ void flash_attn_kernel(const float* __restrict__ q,
                                  const float* __restrict__ k,
                                  const float* __restrict__ v,
                                  const int* __restrict__ idx,
                                  const int* __restrict__ ktmin,
                                  const int* __restrict__ ktmax,
                                  const int* __restrict__ csz,
                                  float* __restrict__ out) {
  __shared__ __align__(16) _Float16 Pbuf[QTS * PPAD];

  const int lane = threadIdx.x;
  const int hi   = lane >> 4;       // wave half (WMMA layout)
  const int n    = lane & 15;       // column index within 16
  const int tile = blockIdx.x;
  const int qt   = tile & (NQT - 1);
  const int h    = (tile >> 7) & (HQN - 1);
  const int b    = tile >> 11;
  const int hkv  = h / GRP;
  const int q0   = qt * QTS;
  const int cs   = csz[0];

  const float scale = rsqrtf((float)DD);
  const int kb = 8 * hi;            // A-layout K-offset base for this half

  // ---- Q tile -> 4 WMMA A operands (16x32 f16 each, covering D=128) ----
  // A layout: lane row = lane&15; halves 0..7 -> d = kb+0..7, 8..15 -> kb+16..23
  const float* qrow = q + (((size_t)b * SEQ + (size_t)(q0 + n)) * HQN + h) * DD;
  v16h A[4];
#pragma unroll
  for (int s = 0; s < 4; ++s) {
    const float* p = qrow + s * 32 + kb;
#pragma unroll
    for (int j = 0; j < 8; ++j) {
      A[s][j]     = (_Float16)(p[j] * scale);
      A[s][8 + j] = (_Float16)(p[16 + j] * scale);
    }
  }

  // ---- per-row block labels + q-tile label range (for tile skipping) ----
  const int* idxb = idx + b * SEQ;
  int idxq[8];
#pragma unroll
  for (int r = 0; r < 8; ++r) idxq[r] = idxb[q0 + r + 8 * hi];
  int lqmin = INT_MAX, lqmax = 0;
#pragma unroll
  for (int r = 0; r < 8; ++r) {
    const int vv = idxq[r];
    if (vv > 0) { lqmin = min(lqmin, vv); lqmax = max(lqmax, vv); }
  }
#pragma unroll
  for (int msk = 1; msk <= 16; msk <<= 1) {   // reduce across whole wave
    lqmin = min(lqmin, __shfl_xor(lqmin, msk, 32));
    lqmax = max(lqmax, __shfl_xor(lqmax, msk, 32));
  }

  const int qc          = q0 / cs;
  const int chunk_start = qc * cs;
  const int q_last      = q0 + QTS - 1;

  // ---- accumulators ----
  v8f   O[8];
  float l[8], mrun[8];
#pragma unroll
  for (int t = 0; t < 8; ++t)
#pragma unroll
    for (int r = 0; r < 8; ++r) O[t][r] = 0.0f;
#pragma unroll
  for (int r = 0; r < 8; ++r) { l[r] = 0.0f; mrun[r] = -__builtin_inff(); }

  const float* kbase = k + ((size_t)b * SEQ * HKVN + hkv) * DD;
  const float* vbase = v + ((size_t)b * SEQ * HKVN + hkv) * DD;
  const int* ktmn = ktmin + b * NKT;
  const int* ktmx = ktmax + b * NKT;

  for (int kt = 0; kt < SEQ; kt += KTS) {
    const int ti = kt >> 5;
    const bool causal_rel = (kt <= q_last) && (kt + KTS - 1 >= chunk_start);
    const int  km = ktmn[ti], kM = ktmx[ti];
    const bool bid_rel = (max(km, lqmin) <= min(kM, lqmax));
    if (!(causal_rel || bid_rel)) continue;   // uniform across wave

    // ---- S = Q * K^T for two 16x16 key column tiles ----
    float Sv[2][8];
#pragma unroll
    for (int nt = 0; nt < 2; ++nt) {
      const int key = kt + nt * 16 + n;       // this lane's key column
      v8f acc;
#pragma unroll
      for (int r = 0; r < 8; ++r) acc[r] = 0.0f;
      // B layout: lane col = key; halves j -> d = 32*s + 8*hi + j, 8+j -> +16 (table)
      const float* krow = kbase + (size_t)key * HKVN * DD + 8 * hi;
#pragma unroll
      for (int s = 0; s < 4; ++s) {
        v16h Bk;
        const float* p = krow + s * 32;
#pragma unroll
        for (int j = 0; j < 8; ++j) { Bk[j] = (_Float16)p[j]; Bk[8 + j] = (_Float16)p[16 + j]; }
        acc = __builtin_amdgcn_wmma_f32_16x16x32_f16(false, A[s], false, Bk,
                                                     (short)0, acc, false, false);
      }
      // ---- elementwise mask (chunk-causal OR same nonzero block label) ----
      const int kc   = key / cs;
      const int idxk = idxb[key];
#pragma unroll
      for (int r = 0; r < 8; ++r) {
        const int  qq = q0 + r + 8 * hi;      // row this lane/VGPR holds
        const bool ok = ((kc == qc) && (key <= qq)) ||
                        ((idxk > 0) && (idxk == idxq[r]));
        Sv[nt][r] = ok ? acc[r] : MASKV;
      }
    }

    // ---- online softmax: row max / rescale / exp / row sum ----
    float mn[8];
#pragma unroll
    for (int r = 0; r < 8; ++r)
      mn[r] = fmaxf(mrun[r], fmaxf(Sv[0][r], Sv[1][r]));
#pragma unroll
    for (int msk = 1; msk <= 8; msk <<= 1)    // within 16-lane half only
#pragma unroll
      for (int r = 0; r < 8; ++r)
        mn[r] = fmaxf(mn[r], __shfl_xor(mn[r], msk, 32));

    float fac[8], rs[8];
#pragma unroll
    for (int r = 0; r < 8; ++r) {
      fac[r]  = __expf(mrun[r] - mn[r]);
      mrun[r] = mn[r];
      const float p0 = __expf(Sv[0][r] - mn[r]);
      const float p1 = __expf(Sv[1][r] - mn[r]);
      Sv[0][r] = p0; Sv[1][r] = p1;
      rs[r] = p0 + p1;
    }
#pragma unroll
    for (int msk = 1; msk <= 8; msk <<= 1)
#pragma unroll
      for (int r = 0; r < 8; ++r) rs[r] += __shfl_xor(rs[r], msk, 32);
#pragma unroll
    for (int r = 0; r < 8; ++r) l[r] = l[r] * fac[r] + rs[r];
#pragma unroll
    for (int t = 0; t < 8; ++t)
#pragma unroll
      for (int r = 0; r < 8; ++r) O[t][r] *= fac[r];

    // ---- P (C layout) -> A layout via LDS transpose ----
    __syncthreads();                          // WAR vs previous iter's reads
#pragma unroll
    for (int nt = 0; nt < 2; ++nt)
#pragma unroll
      for (int r = 0; r < 8; ++r)
        Pbuf[(r + 8 * hi) * PPAD + nt * 16 + n] = (_Float16)Sv[nt][r];
    __syncthreads();
    v16h PA;
    const _Float16* pr = Pbuf + (lane & 15) * PPAD + kb;
#pragma unroll
    for (int j = 0; j < 8; ++j) { PA[j] = pr[j]; PA[8 + j] = pr[16 + j]; }

    // ---- O += P * V  (8 column tiles of 16 over D=128) ----
#pragma unroll
    for (int t = 0; t < 8; ++t) {
      v16h Bv;
      // B layout: lane col = d = 16*t + n; halves j -> key = kt + 8*hi + j, 8+j -> +16
      const float* vcol = vbase + (size_t)(kt + 8 * hi) * HKVN * DD + t * 16 + n;
#pragma unroll
      for (int j = 0; j < 8; ++j) {
        Bv[j] = (_Float16)vcol[(size_t)j * HKVN * DD];
        Bv[8 + j] = (_Float16)vcol[(size_t)(16 + j) * HKVN * DD];
      }
      O[t] = __builtin_amdgcn_wmma_f32_16x16x32_f16(false, PA, false, Bv,
                                                    (short)0, O[t], false, false);
    }
  }

  // ---- epilogue: normalize and store (rows always have >=1 allowed key) ----
  float* obase = out + (((size_t)b * SEQ + q0) * HQN + h) * DD;
#pragma unroll
  for (int r = 0; r < 8; ++r) {
    const float inv = 1.0f / l[r];
    float* orow = obase + (size_t)(r + 8 * hi) * HQN * DD;
#pragma unroll
    for (int t = 0; t < 8; ++t) orow[t * 16 + n] = O[t][r] * inv;
  }
}

// ---------------------------------------------------------------------------
extern "C" void kernel_launch(void* const* d_in, const int* in_sizes, int n_in,
                              void* d_out, int out_size, void* d_ws, size_t ws_size,
                              hipStream_t stream) {
  (void)in_sizes; (void)n_in; (void)out_size; (void)ws_size;
  const float* q  = (const float*)d_in[0];
  const float* k  = (const float*)d_in[1];
  const float* v  = (const float*)d_in[2];
  const int*   bm = (const int*)d_in[3];
  const int*   cs = (const int*)d_in[4];

  int* idx   = (int*)d_ws;            // B*SEQ ints
  int* ktmin = idx + BB * SEQ;        // B*NKT ints
  int* ktmax = ktmin + BB * NKT;      // B*NKT ints

  mask_scan_kernel<<<BB, 256, 0, stream>>>(bm, idx, ktmin, ktmax);
  flash_attn_kernel<<<BB * HQN * NQT, 32, 0, stream>>>(
      q, k, v, idx, ktmin, ktmax, cs, (float*)d_out);
}
